// Image_Text_Attention_7928509628820
// MI455X (gfx1250) — matrix operand order repaired and checked by reading, not compiled, not run on hardware
//
#include <hip/hip_runtime.h>
#include <hip/hip_bf16.h>
#include <math.h>

typedef __attribute__((ext_vector_type(16))) __bf16 v16bf;
typedef __attribute__((ext_vector_type(8)))  __bf16 v8bf;
typedef __attribute__((ext_vector_type(8)))  float  v8f;

#define B_ 64
#define R_ 196
#define L_ 512
#define K_ 768
#define NT_PER_WAVE 6   // 48 n-tiles / 8 waves

__device__ __forceinline__ unsigned short f2bf(float f) {
    unsigned int u = __float_as_uint(f);
    unsigned int r = u + 0x7FFFu + ((u >> 16) & 1u);   // round-to-nearest-even
    return (unsigned short)(r >> 16);
}

__device__ __forceinline__ float wred_sum(float v) {
    #pragma unroll
    for (int o = 16; o > 0; o >>= 1) v += __shfl_xor(v, o, 32);
    return v;
}
__device__ __forceinline__ float wred_max(float v) {
    #pragma unroll
    for (int o = 16; o > 0; o >>= 1) v = fmaxf(v, __shfl_xor(v, o, 32));
    return v;
}

// -------- p / q: wave-per-row dot + tanh --------------------------------
__global__ __launch_bounds__(256)
void pq_kernel(const float* __restrict__ emb, const float* __restrict__ W,
               const float* __restrict__ bias, float* __restrict__ out,
               int nrows, int len) {
    int wid = threadIdx.x >> 5, lane = threadIdx.x & 31;
    int row = blockIdx.x * 8 + wid;
    if (row >= nrows) return;
    const float* rp = emb + (size_t)row * len;
    float acc = 0.f;
    for (int j = lane; j < len; j += 32) acc += rp[j] * W[j];
    acc = wred_sum(acc);
    if (lane == 0) out[row] = tanhf(acc + bias[0]);
}

// -------- s[b,j] = sum_l q[b,l] * V[l,j] --------------------------------
__global__ __launch_bounds__(256)
void s_kernel(const float* __restrict__ q, const float* __restrict__ V,
              float* __restrict__ s) {
    __shared__ float qs[L_];
    int b = blockIdx.x;
    for (int j = threadIdx.x; j < L_; j += 256) qs[j] = q[b * L_ + j];
    __syncthreads();
    int j0 = threadIdx.x, j1 = threadIdx.x + 256;
    float a0 = 0.f, a1 = 0.f;
    for (int l = 0; l < L_; ++l) {
        float ql = qs[l];
        const float* vr = V + (size_t)l * L_;
        a0 += ql * vr[j0];
        a1 += ql * vr[j1];
    }
    s[b * L_ + j0] = a0;
    s[b * L_ + j1] = a1;
}

// -------- transpose+convert: seqT[b][k][l] = bf16(seq[b][l][k]) ---------
// 32x32 tile; packed b32 stores (2 bf16 per store).
__global__ __launch_bounds__(256)
void tconv_kernel(const float* __restrict__ seq, unsigned int* __restrict__ seqT32) {
    __shared__ float tile[32][33];
    int b  = blockIdx.z;
    int k0 = blockIdx.x * 32, l0 = blockIdx.y * 32;
    int tx = threadIdx.x & 31, ty = threadIdx.x >> 5;   // (32, 8)
    const float* src = seq + (size_t)b * L_ * K_;
    #pragma unroll
    for (int i = 0; i < 4; ++i)
        tile[ty + i * 8][tx] = src[(size_t)(l0 + ty + i * 8) * K_ + k0 + tx];
    __syncthreads();
    unsigned int* dst = seqT32 + (size_t)b * K_ * (L_ / 2);
    int lp = threadIdx.x & 15;              // l-pair index 0..15
    #pragma unroll
    for (int i = 0; i < 2; ++i) {
        int kl = (threadIdx.x >> 4) + i * 16;   // k_local 0..31
        unsigned int lo = f2bf(tile[2 * lp][kl]);
        unsigned int hi = f2bf(tile[2 * lp + 1][kl]);
        dst[(size_t)(k0 + kl) * (L_ / 2) + (l0 >> 1) + lp] = lo | (hi << 16);
    }
}

// -------- main: masked softmax of (1+p_r)*s, then W @ seqT via WMMA -----
__global__ __launch_bounds__(256)
void attn_kernel(const float* __restrict__ p, const float* __restrict__ s,
                 const int* __restrict__ mask, const unsigned short* __restrict__ seqT,
                 float* __restrict__ out) {
    __shared__ unsigned short wlds[16][520];   // bf16 softmax weights, padded rows
    int b   = blockIdx.y;
    int r0  = blockIdx.x * 16;
    int wid = threadIdx.x >> 5, lane = threadIdx.x & 31;

    // ---- softmax phase: each wave owns 2 rows of the 16-row tile ----
    #pragma unroll
    for (int rr = 0; rr < 2; ++rr) {
        int rl = wid * 2 + rr;
        int r  = r0 + rl;
        if (r < R_) {
            float c = 1.0f + p[b * R_ + r];
            float vals[16];
            float mx = -3.402823466e38f;
            #pragma unroll
            for (int i = 0; i < 16; ++i) {
                int j = lane + 32 * i;
                float val = (mask[b * L_ + j] == 0) ? -2147483647.0f
                                                    : c * s[b * L_ + j];
                vals[i] = val;
                mx = fmaxf(mx, val);
            }
            mx = wred_max(mx);
            float sum = 0.f;
            #pragma unroll
            for (int i = 0; i < 16; ++i) {
                float e = __expf(vals[i] - mx);
                vals[i] = e;
                sum += e;
            }
            sum = wred_sum(sum);
            float inv = 1.0f / sum;
            #pragma unroll
            for (int i = 0; i < 16; ++i)
                wlds[rl][lane + 32 * i] = f2bf(vals[i] * inv);
        } else {
            #pragma unroll
            for (int i = 0; i < 16; ++i) wlds[rl][lane + 32 * i] = 0;
        }
    }
    __syncthreads();

    // ---- WMMA phase: load A once per k-step, feed 6 accumulators ----
    const int h = lane >> 4, m = lane & 15;
    // B base for tile t: row (wid + 8t)*16 + m of seqT[b], K-offset 16h.
    const __bf16* bbase = (const __bf16*)seqT
        + ((size_t)b * K_ + wid * 16 + m) * L_ + 16 * h;
    const size_t tstride = (size_t)128 * L_;   // 8 n-tiles apart

    v8f acc[NT_PER_WAVE];
    #pragma unroll
    for (int t = 0; t < NT_PER_WAVE; ++t) acc[t] = v8f{};

    #pragma unroll 2
    for (int kk = 0; kk < 16; ++kk) {
        int kb = kk * 32;
        // A fragment (16x32 bf16): lane(m,h) holds K=[kb+16h..+15]
        v8bf alo = *(const v8bf*)&wlds[m][kb + 16 * h];
        v8bf ahi = *(const v8bf*)&wlds[m][kb + 16 * h + 8];
        v16bf a;
        #pragma unroll
        for (int i = 0; i < 8; ++i) { a[i] = alo[i]; a[i + 8] = ahi[i]; }
        #pragma unroll
        for (int t = 0; t < NT_PER_WAVE; ++t) {
            v16bf bf = *(const v16bf*)(bbase + t * tstride + kb);
            acc[t] = __builtin_amdgcn_wmma_f32_16x16x32_bf16(
                false, a, false, bf, (short)0, acc[t], false, false);
        }
    }

    // ---- store: D lane(n=m, h) holds rows h*8+v, col nt*16+n ----
    const float scale = 0.03608439182435161f;  // 1/sqrt(768)
    int mb = h * 8;
    float* obase = out + ((size_t)b * R_ + r0 + mb) * K_ + wid * 16 + m;
    if (r0 + 16 <= R_) {                       // full tile: unconditional stores
        #pragma unroll
        for (int t = 0; t < NT_PER_WAVE; ++t) {
            #pragma unroll
            for (int v = 0; v < 8; ++v)
                obase[(size_t)v * K_ + t * 128] = acc[t][v] * scale;
        }
    } else {                                   // last partial row-tile
        #pragma unroll
        for (int t = 0; t < NT_PER_WAVE; ++t) {
            #pragma unroll
            for (int v = 0; v < 8; ++v) {
                int r = r0 + mb + v;
                if (r < R_) obase[(size_t)v * K_ + t * 128] = acc[t][v] * scale;
            }
        }
    }
}

extern "C" void kernel_launch(void* const* d_in, const int* in_sizes, int n_in,
                              void* d_out, int out_size, void* d_ws, size_t ws_size,
                              hipStream_t stream) {
    const float* image_emb = (const float*)d_in[0];
    const float* seq_emb   = (const float*)d_in[1];
    const int*   mask      = (const int*)d_in[2];
    const float* image_W   = (const float*)d_in[3];
    const float* image_b   = (const float*)d_in[4];
    const float* seq_W     = (const float*)d_in[5];
    const float* seq_b     = (const float*)d_in[6];
    const float* V_weight  = (const float*)d_in[7];
    float* out = (float*)d_out;

    char* ws = (char*)d_ws;
    unsigned short* seqT = (unsigned short*)ws;                // B*K*L bf16 = 50.3 MB
    float* p = (float*)(ws + (size_t)B_ * K_ * L_ * 2);        // B*R
    float* q = p + B_ * R_;                                    // B*L
    float* s = q + B_ * L_;                                    // B*L

    int prows = B_ * R_;
    pq_kernel<<<(prows + 7) / 8, 256, 0, stream>>>(image_emb, image_W, image_b, p, prows, L_);
    int qrows = B_ * L_;
    pq_kernel<<<(qrows + 7) / 8, 256, 0, stream>>>(seq_emb, seq_W, seq_b, q, qrows, K_);
    s_kernel<<<B_, 256, 0, stream>>>(q, V_weight, s);
    tconv_kernel<<<dim3(K_ / 32, L_ / 32, B_), dim3(256), 0, stream>>>(
        seq_emb, (unsigned int*)seqT);
    attn_kernel<<<dim3((R_ + 15) / 16, B_), 256, 0, stream>>>(p, s, mask, seqT, out);
}
